// MultiHeadSelfAttention_50525995270789
// MI455X (gfx1250) — matrix operand order repaired and checked by reading, not compiled, not run on hardware
//
#include <hip/hip_runtime.h>
#include <hip/hip_bf16.h>

// ---------------------------------------------------------------------------
// MultiHeadSelfAttention with banded relative-position bias, CDNA5 (gfx1250)
// bf16 WMMA (v_wmma_f32_16x16x32_bf16), fp32 accumulation, flash softmax,
// TDM (tensor_load_to_lds) double-buffered K/V tiles shared across the block.
// ---------------------------------------------------------------------------

#define B_  4
#define C_  512
#define T_  1024
#define H_  8
#define KC_ 64

typedef unsigned short u16;
typedef unsigned int   u32;
typedef __attribute__((ext_vector_type(16))) __bf16 v16bf;
typedef __attribute__((ext_vector_type(8)))  float  v8f;
typedef __attribute__((ext_vector_type(4)))  u32    u32x4;
typedef __attribute__((ext_vector_type(8)))  int    i32x8;
typedef __attribute__((ext_vector_type(4)))  int    i32x4;
typedef __attribute__((ext_vector_type(2)))  float  f32x2;

#ifndef __has_builtin
#define __has_builtin(x) 0
#endif
// Target builtins are only visible in the device pass; the host pass must
// parse the fallback path (the #error may fire only for gfx1250).
#if defined(__HIP_DEVICE_COMPILE__)
  #if __has_builtin(__builtin_amdgcn_tensor_load_to_lds)
    #define USE_TDM 1
  #else
    #error "gfx1250 device pass: tensor_load_to_lds builtin not available"
  #endif
#else
  #define USE_TDM 0
#endif

union Frag { u32 u[8]; u32x4 q4[2]; v16bf v; };

__device__ __forceinline__ u16 f2bf(float f) {
  u32 u = __builtin_bit_cast(u32, f);
  u32 r = u + 0x7FFFu + ((u >> 16) & 1u);   // round-to-nearest-even
  return (u16)(r >> 16);
}
__device__ __forceinline__ u32 pack2bf(float lo, float hi) {
  return (u32)f2bf(lo) | ((u32)f2bf(hi) << 16);
}
__device__ __forceinline__ float bf2f(u16 h) {
  return __builtin_bit_cast(float, (u32)h << 16);
}
__device__ __forceinline__ v8f wmma_bf16(const Frag& a, const Frag& b, v8f c) {
  return __builtin_amdgcn_wmma_f32_16x16x32_bf16(
      false, a.v, false, b.v, (short)0, c, false, false);
}

// A fragment (16x32, M=lane row): u[j] holds K = k0 + (j>>2)*16 + hi*8 + (j&3)*2
__device__ __forceinline__ Frag load_fragA(const u32* __restrict__ row, int k0, int hi) {
  Frag f;
  const int base = (k0 >> 1) + hi * 4;
  f.q4[0] = *reinterpret_cast<const u32x4*>(row + base);
  f.q4[1] = *reinterpret_cast<const u32x4*>(row + base + 8);
  return f;
}
// B fragment (32x16, N=lane col): u[j] holds K = k0 + (j>>2)*16 + hi*8 + (j&3)*2
__device__ __forceinline__ Frag load_fragB(const u32* __restrict__ row, int k0, int hi) {
  Frag f;
  const int base = (k0 >> 1) + hi * 4;
  f.q4[0] = *reinterpret_cast<const u32x4*>(row + base);
  f.q4[1] = *reinterpret_cast<const u32x4*>(row + base + 8);
  return f;
}

__device__ __forceinline__ float rmax16(float v) {
#pragma unroll
  for (int m = 1; m < 16; m <<= 1) v = fmaxf(v, __shfl_xor(v, m, 16));
  return v;
}
__device__ __forceinline__ float rsum16(float v) {
#pragma unroll
  for (int m = 1; m < 16; m <<= 1) v += __shfl_xor(v, m, 16);
  return v;
}

#if USE_TDM
// 2D Tensor DMA: tile (tl0 x tl1) of a tensor with row stride stride0 (elements,
// 2-byte data) -> LDS at lds_off.  D# packing per CDNA5 ISA §8.3/8.4.
__device__ __forceinline__ void tdm_load_2d(u32 lds_off, const void* gaddr,
                                            u32 td0, u32 td1, u32 tl0, u32 tl1,
                                            u32 stride0) {
  const unsigned long long ga = (unsigned long long)(size_t)gaddr;
  u32x4 g0;
  g0[0] = 1u;                                            // count=1 (user D#)
  g0[1] = lds_off;                                       // lds_addr
  g0[2] = (u32)ga;                                       // global_addr[31:0]
  g0[3] = (u32)((ga >> 32) & 0x01ffffffu) | (2u << 30);  // addr[56:32], type=2
  i32x8 g1;
  g1[0] = (int)(1u << 16);                               // data_size=1 (2 bytes)
  g1[1] = (int)((td0 & 0xffffu) << 16);                  // tensor_dim0 lo16
  g1[2] = (int)((td0 >> 16) | ((td1 & 0xffffu) << 16));  // dim0 hi / dim1 lo
  g1[3] = (int)((td1 >> 16) | (tl0 << 16));              // dim1 hi / tile_dim0
  g1[4] = (int)(tl1 & 0xffffu);                          // tile_dim1, tile_dim2=0
  g1[5] = (int)stride0;                                  // tensor_dim0_stride lo32
  g1[6] = 0;
  g1[7] = 0;
  const i32x4 z4 = {0, 0, 0, 0};
#if defined(__clang_major__) && (__clang_major__ >= 23)
  const i32x8 z8 = {0, 0, 0, 0, 0, 0, 0, 0};
  __builtin_amdgcn_tensor_load_to_lds(g0, g1, z4, z4, z8, 0);
#else
  __builtin_amdgcn_tensor_load_to_lds(g0, g1, z4, z4, 0);
#endif
}
#endif

// ---------------------------------------------------------------------------
// Flash attention per (b,h).  All 8 waves share the same K/V s-chunks ->
// tiles live in LDS, double-buffered, filled by the Tensor Data Mover
// (tensor_load_to_lds + s_wait_tensorcnt) issued by wave 0, one barrier/chunk.
// grid (B*H, T/128), block 256.
// ---------------------------------------------------------------------------
__global__ void __launch_bounds__(256)
flash_attn(const u16* __restrict__ qbf, const u16* __restrict__ kbf,
           const u16* __restrict__ vbf,
           const float* __restrict__ rd, const float* __restrict__ bs,
           const float* __restrict__ embv, u16* __restrict__ aot) {
  __shared__ u16  kbuf[2][32 * KC_];   // K chunk: 32 rows(s) x 64 (kc)
  __shared__ u16  vbuf[2][KC_ * 32];   // V chunk: 64 rows(c) x 32 (s)
  __shared__ u16  pst[8][16 * 32];     // per-wave P staging
  __shared__ float embv_s[9 * KC_];
  for (int i = threadIdx.x; i < 9 * KC_; i += 256) embv_s[i] = embv[i];
  __syncthreads();

  const int wave = threadIdx.x >> 5, lane = threadIdx.x & 31;
  const int laneN = lane & 15, hi = lane >> 4;
  const int bh = blockIdx.x;
  const int t0 = (blockIdx.y * 8 + wave) * 16;

  Frag aq[2];
  {
    const u32* qrow = reinterpret_cast<const u32*>(qbf + ((size_t)bh * T_ + t0 + laneN) * KC_);
    aq[0] = load_fragA(qrow, 0, hi);
    aq[1] = load_fragA(qrow, 32, hi);
  }

  v8f O[4];
#pragma unroll
  for (int cb = 0; cb < 4; ++cb) O[cb] = 0.f;
  float m[8], l[8];
#pragma unroll
  for (int r = 0; r < 8; ++r) { m[r] = -1e30f; l[r] = 0.f; }

#if USE_TDM
  if (wave == 0) {   // preload chunk 0 (TDM: EXEC-ignored, per-wave TENSORcnt)
    tdm_load_2d((u32)(size_t)&kbuf[0][0], kbf + (size_t)bh * T_ * KC_,
                KC_, T_, KC_, 32, KC_);
    tdm_load_2d((u32)(size_t)&vbuf[0][0], vbf + (size_t)bh * KC_ * T_,
                T_, KC_, 32, KC_, T_);
  }
#endif

  for (int sc = 0; sc < T_; sc += 32) {
    const int buf = (sc >> 5) & 1;
#if USE_TDM
    if (wave == 0) __builtin_amdgcn_s_wait_tensorcnt(0);
    __syncthreads();                       // chunk `sc` visible to all waves
    if (wave == 0 && sc + 32 < T_) {       // prefetch next chunk into buf^1
      const int nb = buf ^ 1;
      tdm_load_2d((u32)(size_t)&kbuf[nb][0],
                  kbf + ((size_t)bh * T_ + sc + 32) * KC_, KC_, T_, KC_, 32, KC_);
      tdm_load_2d((u32)(size_t)&vbuf[nb][0],
                  vbf + (size_t)bh * KC_ * T_ + sc + 32, T_, KC_, 32, KC_, T_);
    }
#else
    {  // host-parse / fallback path: cooperative synchronous fill
      const u32* kg = reinterpret_cast<const u32*>(kbf + ((size_t)bh * T_ + sc) * KC_);
      const u32* vg = reinterpret_cast<const u32*>(vbf + (size_t)bh * KC_ * T_ + sc);
      u32* kl = reinterpret_cast<u32*>(&kbuf[buf][0]);
      u32* vl = reinterpret_cast<u32*>(&vbuf[buf][0]);
      for (int i = threadIdx.x; i < 1024; i += 256) kl[i] = kg[i];
      for (int i = threadIdx.x; i < 1024; i += 256)
        vl[i] = vg[(size_t)(i >> 4) * (T_ / 2) + (i & 15)];
      __syncthreads();
    }
#endif

    // ---- S = Q·K^T over 32 columns (K tile from LDS, ds_load_b128) ----
    const u32* klo = reinterpret_cast<const u32*>(&kbuf[buf][laneN * KC_]);
    const u32* khi = reinterpret_cast<const u32*>(&kbuf[buf][(16 + laneN) * KC_]);
    Frag bk[4];
    bk[0] = load_fragB(klo, 0, hi);  bk[1] = load_fragB(klo, 32, hi);
    bk[2] = load_fragB(khi, 0, hi);  bk[3] = load_fragB(khi, 32, hi);
    v8f Slo = 0.f, Shi = 0.f;
    Slo = wmma_bf16(aq[0], bk[0], Slo);
    Slo = wmma_bf16(aq[1], bk[1], Slo);
    Shi = wmma_bf16(aq[0], bk[2], Shi);
    Shi = wmma_bf16(aq[1], bk[3], Shi);

    // ---- banded relative bias where |s-t| <= 4 ----
    if (sc + 31 >= t0 - 4 && sc <= t0 + 19) {
#pragma unroll
      for (int r = 0; r < 8; ++r) {
        const int t = t0 + r + 8 * hi;
        int d = (sc + laneN) - t + 4;
        if (0 <= d && d <= 8) Slo[r] += rd[((size_t)bh * T_ + t) * 9 + d];
        d = (sc + 16 + laneN) - t + 4;
        if (0 <= d && d <= 8) Shi[r] += rd[((size_t)bh * T_ + t) * 9 + d];
      }
    }

    // ---- online softmax ----
#pragma unroll
    for (int r = 0; r < 8; ++r) {
      const float nm = fmaxf(m[r], rmax16(fmaxf(Slo[r], Shi[r])));
      const float resc = __expf(m[r] - nm);
      const float plo = __expf(Slo[r] - nm);
      const float phi = __expf(Shi[r] - nm);
      l[r] = l[r] * resc + rsum16(plo + phi);
      m[r] = nm;
#pragma unroll
      for (int cb = 0; cb < 4; ++cb) O[cb][r] *= resc;
      Slo[r] = plo; Shi[r] = phi;
    }

    // ---- P: C-layout -> A-layout through per-wave LDS ----
#pragma unroll
    for (int r = 0; r < 8; ++r) {
      const int row = r + 8 * hi;
      pst[wave][row * 32 + laneN]      = f2bf(Slo[r]);
      pst[wave][row * 32 + 16 + laneN] = f2bf(Shi[r]);
    }
    asm volatile("s_wait_dscnt 0x0" ::: "memory");
    const Frag ap = load_fragA(
        reinterpret_cast<const u32*>(&pst[wave][laneN * 32]), 0, hi);

    // ---- O += P(16x32)·V(32x64) (V tile from LDS) ----
    Frag bv4[4];
#pragma unroll
    for (int cb = 0; cb < 4; ++cb)
      bv4[cb] = load_fragB(
          reinterpret_cast<const u32*>(&vbuf[buf][(cb * 16 + laneN) * 32]), 0, hi);
#pragma unroll
    for (int cb = 0; cb < 4; ++cb) O[cb] = wmma_bf16(ap, bv4[cb], O[cb]);

#if !USE_TDM
    __syncthreads();   // fallback: protect buf before next refill
#endif
  }

  // ---- epilogue: normalize + banded emb_rel_v, store bf16 [B,T,C] ----
  const int b = bh >> 3, h = bh & 7;
#pragma unroll
  for (int r = 0; r < 8; ++r) {
    const float invl = 1.0f / l[r];
#pragma unroll
    for (int cb = 0; cb < 4; ++cb) O[cb][r] *= invl;
    const int t = t0 + r + 8 * hi;
    for (int d = 0; d < 9; ++d) {
      const int s = t + d - 4;
      if (s < 0 || s >= T_) continue;
      const float p = __expf(bs[((size_t)bh * T_ + t) * 9 + d] - m[r]) * invl;
#pragma unroll
      for (int cb = 0; cb < 4; ++cb)
        O[cb][r] += p * embv_s[d * KC_ + cb * 16 + laneN];
    }
#pragma unroll
    for (int cb = 0; cb < 4; ++cb) {
      const int c = h * KC_ + cb * 16 + laneN;
      aot[((size_t)b * T_ + t) * C_ + c] = f2bf(O[cb][r]);
    }
  }
}

// ---------------------------------------------------------------------------
// Prep 1: weights f32 -> bf16 (wq|wk|wv|wo packed consecutively in ws).
// ---------------------------------------------------------------------------
__global__ void __launch_bounds__(256)
cvt_w(const float* __restrict__ s0, const float* __restrict__ s1,
      const float* __restrict__ s2, const float* __restrict__ s3,
      u16* __restrict__ dst) {
  const float* s = (blockIdx.z == 0) ? s0 : (blockIdx.z == 1) ? s1
                 : (blockIdx.z == 2) ? s2 : s3;
  const int i = blockIdx.x * 256 + threadIdx.x;
  const f32x2 w = reinterpret_cast<const f32x2*>(s)[i];
  reinterpret_cast<u32*>(dst + (size_t)blockIdx.z * C_ * C_)[i] = pack2bf(w[0], w[1]);
}

// ---------------------------------------------------------------------------
// Prep 2: x [B,C,T] f32 -> xt [B,T,C] bf16 (LDS 32x32 tile transpose).
// ---------------------------------------------------------------------------
__global__ void __launch_bounds__(256)
cvt_x(const float* __restrict__ x, u16* __restrict__ xt) {
  __shared__ float tile[32][33];
  const int b = blockIdx.z, c0 = blockIdx.y * 32, t0 = blockIdx.x * 32;
  const int tx = threadIdx.x & 31, ty = threadIdx.x >> 5;
#pragma unroll
  for (int k = 0; k < 4; ++k)
    tile[ty + 8 * k][tx] = x[((size_t)b * C_ + c0 + ty + 8 * k) * T_ + t0 + tx];
  __syncthreads();
#pragma unroll
  for (int k = 0; k < 4; ++k)
    xt[((size_t)b * T_ + t0 + ty + 8 * k) * C_ + c0 + tx] = f2bf(tile[tx][ty + 8 * k]);
}

// ---------------------------------------------------------------------------
// q/k projection: M=t, N=o so [BH,T,KC] stores coalesce.  q scaled 1/sqrt(KC).
// grid (T/128, C/64, 2B), block 256; wave: 16(t) x 64(o).
// ---------------------------------------------------------------------------
__global__ void __launch_bounds__(256)
qk_proj(const u16* __restrict__ xt, const u16* __restrict__ wbf,
        const float* __restrict__ bq, const float* __restrict__ bk,
        u16* __restrict__ qbf, u16* __restrict__ kbf) {
  const int wave = threadIdx.x >> 5, lane = threadIdx.x & 31;
  const int laneN = lane & 15, hi = lane >> 4;
  const int mat = blockIdx.z & 1, b = blockIdx.z >> 1;
  const int t0 = (blockIdx.x * 8 + wave) * 16;
  const int o0 = blockIdx.y * 64;

  const u16* W = wbf + (size_t)mat * C_ * C_;
  const float* bias = mat ? bk : bq;
  const u32* arow = reinterpret_cast<const u32*>(xt + ((size_t)b * T_ + t0 + laneN) * C_);
  const u32* brow[4];
#pragma unroll
  for (int nt = 0; nt < 4; ++nt)
    brow[nt] = reinterpret_cast<const u32*>(W + (size_t)(o0 + nt * 16 + laneN) * C_);

  v8f acc[4];
#pragma unroll
  for (int nt = 0; nt < 4; ++nt) acc[nt] = 0.f;

  for (int i0 = 0; i0 < C_; i0 += 32) {
    const Frag a = load_fragA(arow, i0, hi);
    Frag bm[4];
#pragma unroll
    for (int nt = 0; nt < 4; ++nt) bm[nt] = load_fragB(brow[nt], i0, hi);
#pragma unroll
    for (int nt = 0; nt < 4; ++nt) acc[nt] = wmma_bf16(a, bm[nt], acc[nt]);
  }

  const float scale = mat ? 1.0f : 0.125f;
  const int h = o0 >> 6;
  const size_t bh = (size_t)(b * H_ + h);
  u16* dst = mat ? kbf : qbf;
#pragma unroll
  for (int nt = 0; nt < 4; ++nt) {
#pragma unroll
    for (int r = 0; r < 8; ++r) {
      const int t = t0 + r + 8 * hi;
      const int o = o0 + nt * 16 + laneN;
      dst[(bh * T_ + t) * KC_ + nt * 16 + laneN] = f2bf((acc[nt][r] + bias[o]) * scale);
    }
  }
}

// ---------------------------------------------------------------------------
// v projection: M=o(kc), N=t so [BH,KC,T] stores coalesce.
// grid (C/128, T/64, B), block 256; wave: 16(o) x 64(t).
// ---------------------------------------------------------------------------
__global__ void __launch_bounds__(256)
v_proj(const u16* __restrict__ xt, const u16* __restrict__ wbf,
       const float* __restrict__ bv, u16* __restrict__ vbf) {
  const int wave = threadIdx.x >> 5, lane = threadIdx.x & 31;
  const int laneN = lane & 15, hi = lane >> 4;
  const int b = blockIdx.z;
  const int o0 = (blockIdx.x * 8 + wave) * 16;
  const int t0 = blockIdx.y * 64;

  const u16* W = wbf + (size_t)2 * C_ * C_;
  const u32* arow = reinterpret_cast<const u32*>(W + (size_t)(o0 + laneN) * C_);
  const u32* brow[4];
#pragma unroll
  for (int nt = 0; nt < 4; ++nt)
    brow[nt] = reinterpret_cast<const u32*>(xt + ((size_t)b * T_ + t0 + nt * 16 + laneN) * C_);

  v8f acc[4];
#pragma unroll
  for (int nt = 0; nt < 4; ++nt) acc[nt] = 0.f;

  for (int i0 = 0; i0 < C_; i0 += 32) {
    const Frag a = load_fragA(arow, i0, hi);
    Frag bm[4];
#pragma unroll
    for (int nt = 0; nt < 4; ++nt) bm[nt] = load_fragB(brow[nt], i0, hi);
#pragma unroll
    for (int nt = 0; nt < 4; ++nt) acc[nt] = wmma_bf16(a, bm[nt], acc[nt]);
  }

#pragma unroll
  for (int nt = 0; nt < 4; ++nt) {
#pragma unroll
    for (int r = 0; r < 8; ++r) {
      const int o = o0 + r + 8 * hi;
      const int tc = t0 + nt * 16 + laneN;
      const size_t bh = (size_t)(b * H_ + (o >> 6));
      vbf[(bh * KC_ + (o & 63)) * T_ + tc] = f2bf(acc[nt][r] + bv[o]);
    }
  }
}

// ---------------------------------------------------------------------------
// Banded relative terms: rd[t,d] = q[t]·emb_k[d];  bs[t,d] = q[t]·k[t+d-4]+rd.
// ---------------------------------------------------------------------------
__global__ void __launch_bounds__(256)
band_kernel(const u16* __restrict__ qbf, const u16* __restrict__ kbf,
            const float* __restrict__ embk,
            float* __restrict__ rd, float* __restrict__ bs) {
  const int idx = blockIdx.x * 256 + threadIdx.x;
  const int t  = idx & (T_ - 1);
  const int bh = idx >> 10;

  float qv[KC_];
  const u32* qp = reinterpret_cast<const u32*>(qbf + (size_t)idx * KC_);
#pragma unroll
  for (int i = 0; i < KC_ / 2; ++i) {
    const u32 w = qp[i];
    qv[2 * i]     = bf2f((u16)(w & 0xffff));
    qv[2 * i + 1] = bf2f((u16)(w >> 16));
  }

#pragma unroll
  for (int d = 0; d < 9; ++d) {
    float relp = 0.f;
#pragma unroll
    for (int c = 0; c < KC_; ++c) relp += qv[c] * embk[d * KC_ + c];
    rd[(size_t)idx * 9 + d] = relp;

    const int s = t + d - 4;
    float out = -1e30f;
    if (0 <= s && s < T_) {
      const u32* kp = reinterpret_cast<const u32*>(kbf + ((size_t)bh * T_ + s) * KC_);
      float dot = 0.f;
#pragma unroll
      for (int i = 0; i < KC_ / 2; ++i) {
        const u32 w = kp[i];
        dot += qv[2 * i] * bf2f((u16)(w & 0xffff));
        dot += qv[2 * i + 1] * bf2f((u16)(w >> 16));
      }
      out = relp + dot;
    }
    bs[(size_t)idx * 9 + d] = out;
  }
}

// ---------------------------------------------------------------------------
// Output projection: out[b,o,t] = wo[o,:]·ao[b,:,t] + bo[o], f32 result.
// ---------------------------------------------------------------------------
__global__ void __launch_bounds__(256)
out_proj(const u16* __restrict__ aot, const u16* __restrict__ wbf,
         const float* __restrict__ bo, float* __restrict__ out) {
  const int wave = threadIdx.x >> 5, lane = threadIdx.x & 31;
  const int laneN = lane & 15, hi = lane >> 4;
  const int b = blockIdx.z;
  const int o0 = (blockIdx.x * 8 + wave) * 16;
  const int t0 = blockIdx.y * 64;

  const u16* W = wbf + (size_t)3 * C_ * C_;
  const u32* arow = reinterpret_cast<const u32*>(W + (size_t)(o0 + laneN) * C_);
  const u32* brow[4];
#pragma unroll
  for (int nt = 0; nt < 4; ++nt)
    brow[nt] = reinterpret_cast<const u32*>(aot + ((size_t)b * T_ + t0 + nt * 16 + laneN) * C_);

  v8f acc[4];
#pragma unroll
  for (int nt = 0; nt < 4; ++nt) acc[nt] = 0.f;

  for (int i0 = 0; i0 < C_; i0 += 32) {
    const Frag a = load_fragA(arow, i0, hi);
    Frag bm[4];
#pragma unroll
    for (int nt = 0; nt < 4; ++nt) bm[nt] = load_fragB(brow[nt], i0, hi);
#pragma unroll
    for (int nt = 0; nt < 4; ++nt) acc[nt] = wmma_bf16(a, bm[nt], acc[nt]);
  }

#pragma unroll
  for (int nt = 0; nt < 4; ++nt) {
    const int tc = t0 + nt * 16 + laneN;
#pragma unroll
    for (int r = 0; r < 8; ++r) {
      const int o = o0 + r + 8 * hi;
      out[((size_t)b * C_ + o) * T_ + tc] = acc[nt][r] + bo[o];
    }
  }
}

// ---------------------------------------------------------------------------
extern "C" void kernel_launch(void* const* d_in, const int* in_sizes, int n_in,
                              void* d_out, int out_size, void* d_ws, size_t ws_size,
                              hipStream_t stream) {
  const float* x    = (const float*)d_in[0];
  const float* wq   = (const float*)d_in[1];
  const float* bq   = (const float*)d_in[2];
  const float* wk   = (const float*)d_in[3];
  const float* bk   = (const float*)d_in[4];
  const float* wv   = (const float*)d_in[5];
  const float* bv   = (const float*)d_in[6];
  const float* wo   = (const float*)d_in[7];
  const float* bo   = (const float*)d_in[8];
  const float* embk = (const float*)d_in[9];
  const float* embv = (const float*)d_in[10];

  char* p = (char*)d_ws;
  u16* wbf = (u16*)p;    p += (size_t)4 * C_ * C_ * sizeof(u16);
  u16* xt  = (u16*)p;    p += (size_t)B_ * T_ * C_ * sizeof(u16);
  u16* qbf = (u16*)p;    p += (size_t)B_ * H_ * T_ * KC_ * sizeof(u16);
  u16* kbf = (u16*)p;    p += (size_t)B_ * H_ * T_ * KC_ * sizeof(u16);
  u16* vbf = (u16*)p;    p += (size_t)B_ * H_ * T_ * KC_ * sizeof(u16);
  float* rd = (float*)p; p += (size_t)B_ * H_ * T_ * 9 * sizeof(float);
  float* bs = (float*)p; p += (size_t)B_ * H_ * T_ * 9 * sizeof(float);
  u16* aot = (u16*)p;    p += (size_t)B_ * T_ * C_ * sizeof(u16);

  cvt_w<<<dim3(C_ * C_ / 512, 1, 4), 256, 0, stream>>>(wq, wk, wv, wo, wbf);
  cvt_x<<<dim3(T_ / 32, C_ / 32, B_), 256, 0, stream>>>(x, xt);
  qk_proj<<<dim3(T_ / 128, C_ / 64, 2 * B_), 256, 0, stream>>>(xt, wbf, bq, bk, qbf, kbf);
  v_proj<<<dim3(C_ / 128, T_ / 64, B_), 256, 0, stream>>>(xt, wbf, bv, vbf);
  band_kernel<<<(B_ * H_ * T_) / 256, 256, 0, stream>>>(qbf, kbf, embk, rd, bs);
  flash_attn<<<dim3(B_ * H_, T_ / 128), 256, 0, stream>>>(qbf, kbf, vbf, rd, bs, embv, aot);
  out_proj<<<dim3(C_ / 128, T_ / 64, B_), 256, 0, stream>>>(aot, wbf, bo, (float*)d_out);
}
